// BiLSTM_CRF_20151986552956
// MI455X (gfx1250) — matrix operand order repaired and checked by reading, not compiled, not run on hardware
//
#include <hip/hip_runtime.h>
#include <hip/hip_bf16.h>

#define EMB 256
#define HID 512
#define HD 256
#define TAGS 11
#define START_TAG 9
#define STOP_TAG 10
#define BB 64
#define TSTEPS 512
#define NEGV (-10000.0f)

typedef __attribute__((ext_vector_type(16))) __bf16 v16bf;
typedef __attribute__((ext_vector_type(8)))  float  v8f;

union Frag { unsigned int u[8]; uint4 q[2]; v16bf bf; };

__device__ inline unsigned short f2bf(float f) {
    unsigned int u = __float_as_uint(f);
    unsigned int r = (u + 0x7FFFu + ((u >> 16) & 1u)) >> 16;
    return (unsigned short)r;
}
__device__ inline float bf2f(unsigned short b) {
    return __uint_as_float(((unsigned int)b) << 16);
}
__device__ inline unsigned int pack2bf(float lo, float hi) {
    return (unsigned int)f2bf(lo) | ((unsigned int)f2bf(hi) << 16);
}
__device__ inline float sigmoidf_(float x) { return 1.0f / (1.0f + __expf(-x)); }

// ---------------------------------------------------------------------------
// Prep: combined recurrent weights -> bf16 row-major [dir][n=1024][k=512]
//       (k<256 = Wih, k>=256 = Whh), plus padded Wtag bf16 [16][512].
// ---------------------------------------------------------------------------
__global__ void prep_weights_kernel(const float* __restrict__ Wih_f, const float* __restrict__ Whh_f,
                                    const float* __restrict__ Wih_b, const float* __restrict__ Whh_b,
                                    const float* __restrict__ Wtag,
                                    unsigned short* __restrict__ Wbf,
                                    unsigned short* __restrict__ Wtagbf) {
    int idx = blockIdx.x * blockDim.x + threadIdx.x;
    if (idx < 2 * 1024 * 512) {
        int dir = idx >> 19;
        int n   = (idx >> 9) & 1023;
        int k   = idx & 511;
        const float* Wih = dir ? Wih_b : Wih_f;
        const float* Whh = dir ? Whh_b : Whh_f;
        float v = (k < HD) ? Wih[n * EMB + k] : Whh[n * HD + (k - HD)];
        Wbf[idx] = f2bf(v);
    }
    if (idx < 16 * 512) {
        int n = idx >> 9, k = idx & 511;
        float v = (n < TAGS) ? Wtag[n * HID + k] : 0.0f;
        Wtagbf[idx] = f2bf(v);
    }
}

// ---------------------------------------------------------------------------
// Embedding gather: Xbf[(t*64+b)][e] = bf16(embed[sentence[b][t]][e])
// ---------------------------------------------------------------------------
__global__ void embed_gather_kernel(const int* __restrict__ sentence,
                                    const float* __restrict__ embed,
                                    unsigned short* __restrict__ Xbf) {
    long idx = (long)blockIdx.x * blockDim.x + threadIdx.x;
    if (idx >= (long)TSTEPS * BB * EMB) return;
    int e   = (int)(idx & (EMB - 1));
    int row = (int)(idx >> 8);        // t*64 + b
    int t = row >> 6, b = row & 63;
    int tok = sentence[b * TSTEPS + t];
    Xbf[idx] = f2bf(embed[(long)tok * EMB + e]);
}

// ---------------------------------------------------------------------------
// Recurrent BiLSTM, spread over 4 WGPs per direction (8 blocks total).
// Block blk: dir = blk>>2, sub = blk&3 owns hidden tiles [sub*4, sub*4+4).
// Each of 16 waves owns one (ht, mt) unit: 4 gate accumulators (32 VGPRs),
// c state in VGPRs. h exchanged between the 4 blocks of a direction via a
// ping-pong global buffer in A-fragment layout; one atomic arrive/spin
// barrier (agent scope) per step. waves_per_eu(4): one block per WGP, so
// allow the full file/4 VGPR budget -> no scratch spills in the hot loop.
// ---------------------------------------------------------------------------
__global__ __launch_bounds__(512)
__attribute__((amdgpu_waves_per_eu(4)))
void lstm_kernel(
    const unsigned short* __restrict__ Xbf, const unsigned short* __restrict__ Wbf,
    const float* __restrict__ b_f, const float* __restrict__ b_b,
    const float* __restrict__ h0f, const float* __restrict__ c0f,
    const float* __restrict__ h0b, const float* __restrict__ c0b,
    const float* __restrict__ mask, unsigned short* __restrict__ H,
    unsigned int* __restrict__ Hx,   // [dir][2][8192] dwords, A-frag layout
    int* __restrict__ barcnt)        // [dir*32]
{
    const int blk  = blockIdx.x;
    const int dir  = blk >> 2;
    const int sub  = blk & 3;            // hidden-tile group of this block
    const int tid  = threadIdx.x;
    const int wave = tid >> 5;
    const int lane = tid & 31;
    const int ht   = sub * 4 + (wave >> 2);   // hidden tile 0..15
    const int mt   = wave & 3;                // batch tile 0..3
    const int nloc = lane & 15;
    const int kh   = lane >> 4;
    const int hidx = ht * 16 + nloc;

    const float* bias = dir ? b_b : b_f;
    const float* h0   = dir ? h0b : h0f;
    const float* c0   = dir ? c0b : c0f;
    unsigned int* hx0 = Hx + (size_t)dir * 2 * 8192;        // buffer 0
    unsigned int* hx1 = hx0 + 8192;                          // buffer 1
    int* cnt = barcnt + dir * 32;

    // Loop-invariant W gate-row fragment base pointers (dword units).
    const unsigned int* Wd = (const unsigned int*)(Wbf + (size_t)dir * 1024 * 512);
    const unsigned int* wg0 = Wd + (size_t)(0 * HD + hidx) * 256 + 8 * kh;
    const unsigned int* wg1 = Wd + (size_t)(1 * HD + hidx) * 256 + 8 * kh;
    const unsigned int* wg2 = Wd + (size_t)(2 * HD + hidx) * 256 + 8 * kh;
    const unsigned int* wg3 = Wd + (size_t)(3 * HD + hidx) * 256 + 8 * kh;

    const float bi  = bias[0 * HD + hidx];
    const float bff = bias[1 * HD + hidx];
    const float bg  = bias[2 * HD + hidx];
    const float bo  = bias[3 * HD + hidx];

    // c state in registers: element (b = mt*16 + r + 8*kh, hidx)
    float creg[8];
#pragma unroll
    for (int r = 0; r < 8; ++r) {
        int b = mt * 16 + r + 8 * kh;
        creg[r] = c0[b * HD + hidx];
    }

    // Init buffer 0 with h0 (this block fills its own kc pair {sub*2, sub*2+1})
    for (int d0 = tid; d0 < 4 * 2 * 32 * 8; d0 += 512) {
        int v   = d0 & 7;
        int ln  = (d0 >> 3) & 31;
        int kcl = (d0 >> 8) & 1;
        int mtt = d0 >> 9;
        int kc  = sub * 2 + kcl;
        int b   = mtt * 16 + (ln & 15);
        int k   = kc * 32 + 16 * (ln >> 4) + 2 * v;
        hx0[(mtt * 8 + kc) * 256 + ln * 8 + v] = pack2bf(h0[b * HD + k], h0[b * HD + k + 1]);
    }

    // inter-block barrier: arrive + wait (agent scope)
    int expected = 4;
    {
        __threadfence();
        __syncthreads();
        if (tid == 0) {
            __hip_atomic_fetch_add(cnt, 1, __ATOMIC_ACQ_REL, __HIP_MEMORY_SCOPE_AGENT);
            while (__hip_atomic_load(cnt, __ATOMIC_ACQUIRE, __HIP_MEMORY_SCOPE_AGENT) < expected)
                __builtin_amdgcn_s_sleep(1);
        }
        __syncthreads();
    }

    for (int step = 0; step < TSTEPS; ++step) {
        const int t = dir ? (TSTEPS - 1 - step) : step;
        const unsigned int* cur = (step & 1) ? hx1 : hx0;   // h at this step
        unsigned int*       nxt = (step & 1) ? hx0 : hx1;   // h for next step

        v8f acc[4];
#pragma unroll
        for (int g = 0; g < 4; ++g) {
            v8f z = {0.f,0.f,0.f,0.f,0.f,0.f,0.f,0.f};
            acc[g] = z;
        }

        // A-frag base for x_t (dword units); +8*kh selects this lane's K-half
        const unsigned int* xrow =
            (const unsigned int*)Xbf + ((size_t)(t * BB + mt * 16 + nloc)) * 128 + 8 * kh;
        // A-frag base for h (already frag-ordered): 8 consecutive dwords/lane
        const unsigned int* hrow = cur + mt * 2048 + lane * 8;

        // ---- K-chunks 0..7 : input contribution ----
#pragma unroll 2
        for (int kc = 0; kc < 8; ++kc) {
            Frag a;
            const uint4* s = (const uint4*)(xrow + kc * 16);
            a.q[0] = s[0];
            a.q[1] = s[1];
            Frag w0, w1, w2, w3;
            w0.q[0] = ((const uint4*)(wg0 + kc * 16))[0];
            w0.q[1] = ((const uint4*)(wg0 + kc * 16))[1];
            w1.q[0] = ((const uint4*)(wg1 + kc * 16))[0];
            w1.q[1] = ((const uint4*)(wg1 + kc * 16))[1];
            w2.q[0] = ((const uint4*)(wg2 + kc * 16))[0];
            w2.q[1] = ((const uint4*)(wg2 + kc * 16))[1];
            w3.q[0] = ((const uint4*)(wg3 + kc * 16))[0];
            w3.q[1] = ((const uint4*)(wg3 + kc * 16))[1];
            acc[0] = __builtin_amdgcn_wmma_f32_16x16x32_bf16(false, a.bf, false, w0.bf, (short)0, acc[0], false, false);
            acc[1] = __builtin_amdgcn_wmma_f32_16x16x32_bf16(false, a.bf, false, w1.bf, (short)0, acc[1], false, false);
            acc[2] = __builtin_amdgcn_wmma_f32_16x16x32_bf16(false, a.bf, false, w2.bf, (short)0, acc[2], false, false);
            acc[3] = __builtin_amdgcn_wmma_f32_16x16x32_bf16(false, a.bf, false, w3.bf, (short)0, acc[3], false, false);
        }
        // ---- K-chunks 8..15 : recurrent contribution ----
#pragma unroll 2
        for (int kc = 8; kc < 16; ++kc) {
            Frag a;
            const uint4* s = (const uint4*)(hrow + (kc - 8) * 256);
            a.q[0] = s[0];
            a.q[1] = s[1];
            Frag w0, w1, w2, w3;
            w0.q[0] = ((const uint4*)(wg0 + kc * 16))[0];
            w0.q[1] = ((const uint4*)(wg0 + kc * 16))[1];
            w1.q[0] = ((const uint4*)(wg1 + kc * 16))[0];
            w1.q[1] = ((const uint4*)(wg1 + kc * 16))[1];
            w2.q[0] = ((const uint4*)(wg2 + kc * 16))[0];
            w2.q[1] = ((const uint4*)(wg2 + kc * 16))[1];
            w3.q[0] = ((const uint4*)(wg3 + kc * 16))[0];
            w3.q[1] = ((const uint4*)(wg3 + kc * 16))[1];
            acc[0] = __builtin_amdgcn_wmma_f32_16x16x32_bf16(false, a.bf, false, w0.bf, (short)0, acc[0], false, false);
            acc[1] = __builtin_amdgcn_wmma_f32_16x16x32_bf16(false, a.bf, false, w1.bf, (short)0, acc[1], false, false);
            acc[2] = __builtin_amdgcn_wmma_f32_16x16x32_bf16(false, a.bf, false, w2.bf, (short)0, acc[2], false, false);
            acc[3] = __builtin_amdgcn_wmma_f32_16x16x32_bf16(false, a.bf, false, w3.bf, (short)0, acc[3], false, false);
        }

        // Activations + state update (writes go to the OTHER buffer: no hazard)
        const unsigned short* cur16 = (const unsigned short*)cur;
        unsigned short*       nxt16 = (unsigned short*)nxt;
#pragma unroll
        for (int r = 0; r < 8; ++r) {
            int   b    = mt * 16 + r + 8 * kh;
            float iv   = sigmoidf_(acc[0][r] + bi);
            float fv   = sigmoidf_(acc[1][r] + bff);
            float gv   = tanhf(acc[2][r] + bg);
            float ov   = sigmoidf_(acc[3][r] + bo);
            float cn   = fv * creg[r] + iv * gv;
            float hn   = ov * tanhf(cn);
            float mval = mask[t * BB + b];

            // A-frag address of element (b, hidx)
            int kk     = hidx & 31;
            int lane_a = (b & 15) + 16 * ((kk >> 4) & 1);
            int v      = (kk >> 1) & 7;
            int d      = (mt * 8 + (hidx >> 5)) * 256 + lane_a * 8 + v;
            int i16    = d * 2 + (kk & 1);

            float hold = bf2f(cur16[i16]);
            float hsel = mval * hn + (1.0f - mval) * hold;
            creg[r]    = mval * cn + (1.0f - mval) * creg[r];

            unsigned short hb = f2bf(hsel);
            nxt16[i16] = hb;
            H[((size_t)(t * BB + b)) * HID + dir * HD + hidx] = hb;
        }

        // barrier: everyone finished step (reads of cur + writes of nxt)
        expected += 4;
        __threadfence();
        __syncthreads();
        if (tid == 0) {
            __hip_atomic_fetch_add(cnt, 1, __ATOMIC_ACQ_REL, __HIP_MEMORY_SCOPE_AGENT);
            while (__hip_atomic_load(cnt, __ATOMIC_ACQUIRE, __HIP_MEMORY_SCOPE_AGENT) < expected)
                __builtin_amdgcn_s_sleep(1);
        }
        __syncthreads();
    }
}

// ---------------------------------------------------------------------------
// feats = (H @ Wtag^T + btag) * mask ; N padded 11->16. One 16x16 tile / wave.
// ---------------------------------------------------------------------------
__global__ __launch_bounds__(256) void feats_kernel(
    const unsigned short* __restrict__ H, const unsigned short* __restrict__ Wtagbf,
    const float* __restrict__ btag, const float* __restrict__ mask,
    float* __restrict__ feats)
{
    const int wave  = threadIdx.x >> 5;
    const int lane  = threadIdx.x & 31;
    const int mtile = blockIdx.x * 8 + wave;   // 2048 row-tiles of 16
    const int nloc  = lane & 15;
    const int kh    = lane >> 4;

    const unsigned int* arow =
        (const unsigned int*)H + ((size_t)(mtile * 16 + nloc)) * 256 + 8 * kh;
    const unsigned int* wrow =
        (const unsigned int*)Wtagbf + (size_t)nloc * 256 + 8 * kh;

    v8f acc = {0.f,0.f,0.f,0.f,0.f,0.f,0.f,0.f};
#pragma unroll 2
    for (int kc = 0; kc < 16; ++kc) {
        Frag a, w;
        const uint4* s = (const uint4*)(arow + kc * 32);
        a.q[0] = s[0];
        a.q[1] = s[1];
        w.q[0] = ((const uint4*)(wrow + kc * 16))[0];
        w.q[1] = ((const uint4*)(wrow + kc * 16))[1];
        acc = __builtin_amdgcn_wmma_f32_16x16x32_bf16(false, a.bf, false, w.bf,
                                                      (short)0, acc, false, false);
    }
    float bt = (nloc < TAGS) ? btag[nloc] : 0.0f;
#pragma unroll
    for (int r = 0; r < 8; ++r) {
        int row = mtile * 16 + r + 8 * kh;
        float mval = mask[row];
        feats[(size_t)row * 16 + nloc] = (acc[r] + bt) * mval;
    }
}

// ---------------------------------------------------------------------------
// Viterbi + backtrack. One block of 64x11 = 704 threads (22 wave32).
// ---------------------------------------------------------------------------
__global__ __launch_bounds__(704) void viterbi_kernel(
    const float* __restrict__ feats, const float* __restrict__ mask,
    const float* __restrict__ trans, int* __restrict__ ptrs,
    float* __restrict__ out)
{
    __shared__ float sc[BB][TAGS];
    __shared__ float tr[TAGS * TAGS];
    __shared__ float trStop[TAGS];

    const int tid = threadIdx.x;
    const int b = tid / TAGS, j = tid % TAGS;

    if (tid < TAGS * TAGS) tr[tid] = trans[tid];
    if (tid < TAGS) trStop[tid] = trans[STOP_TAG * TAGS + tid];
    sc[b][j] = (j == START_TAG) ? 0.0f : NEGV;
    __syncthreads();

    for (int t = 0; t < TSTEPS; ++t) {
        float best = sc[b][0] + tr[j * TAGS + 0];
        int   arg  = 0;
#pragma unroll
        for (int k = 1; k < TAGS; ++k) {
            float v = sc[b][k] + tr[j * TAGS + k];
            if (v > best) { best = v; arg = k; }
        }
        ptrs[(t * BB + b) * 16 + j] = arg;
        float news = best + feats[(size_t)(t * BB + b) * 16 + j];
        float mval = mask[t * BB + b];
        __syncthreads();
        if (mval > 0.0f) sc[b][j] = news;
        __syncthreads();
    }

    if (j == 0) {
        float best = sc[b][0] + trStop[0];
        int   bt   = 0;
#pragma unroll
        for (int k = 1; k < TAGS; ++k) {
            float v = sc[b][k] + trStop[k];
            if (v > best) { best = v; bt = k; }
        }
        out[BB * TSTEPS + b] = best;
        int tag = bt;
        for (int t = TSTEPS - 1; t >= 0; --t) {
            out[(size_t)b * TSTEPS + t] = (float)tag;
            tag = ptrs[(t * BB + b) * 16 + tag];
        }
    }
}

// ---------------------------------------------------------------------------
extern "C" void kernel_launch(void* const* d_in, const int* in_sizes, int n_in,
                              void* d_out, int out_size, void* d_ws, size_t ws_size,
                              hipStream_t stream) {
    const int*   sentence = (const int*)d_in[0];
    const float* mask     = (const float*)d_in[1];
    const float* embed    = (const float*)d_in[2];
    const float* Wih_f    = (const float*)d_in[3];
    const float* Whh_f    = (const float*)d_in[4];
    const float* b_f      = (const float*)d_in[5];
    const float* Wih_b    = (const float*)d_in[6];
    const float* Whh_b    = (const float*)d_in[7];
    const float* b_b      = (const float*)d_in[8];
    const float* h0f      = (const float*)d_in[9];
    const float* c0f      = (const float*)d_in[10];
    const float* h0b      = (const float*)d_in[11];
    const float* c0b      = (const float*)d_in[12];
    const float* Wtag     = (const float*)d_in[13];
    const float* btag     = (const float*)d_in[14];
    const float* trans    = (const float*)d_in[15];

    char* ws = (char*)d_ws;
    unsigned short* Wbf    = (unsigned short*)ws; ws += (size_t)2 * 1024 * 512 * 2;
    unsigned short* Wtagbf = (unsigned short*)ws; ws += (size_t)16 * 512 * 2;
    unsigned short* Xbf    = (unsigned short*)ws; ws += (size_t)TSTEPS * BB * EMB * 2;
    unsigned short* H      = (unsigned short*)ws; ws += (size_t)TSTEPS * BB * HID * 2;
    float*          feats  = (float*)ws;          ws += (size_t)TSTEPS * BB * 16 * 4;
    int*            ptrs   = (int*)ws;            ws += (size_t)TSTEPS * BB * 16 * 4;
    unsigned int*   Hx     = (unsigned int*)ws;   ws += (size_t)2 * 2 * 8192 * 4;
    int*            barcnt = (int*)ws;            ws += 64 * sizeof(int);

    hipMemsetAsync(barcnt, 0, 64 * sizeof(int), stream);

    prep_weights_kernel<<<(2 * 1024 * 512 + 255) / 256, 256, 0, stream>>>(
        Wih_f, Whh_f, Wih_b, Whh_b, Wtag, Wbf, Wtagbf);
    embed_gather_kernel<<<(TSTEPS * BB * EMB + 255) / 256, 256, 0, stream>>>(
        sentence, embed, Xbf);
    lstm_kernel<<<8, 512, 0, stream>>>(Xbf, Wbf, b_f, b_b, h0f, c0f, h0b, c0b,
                                       mask, H, Hx, barcnt);
    feats_kernel<<<256, 256, 0, stream>>>(H, Wtagbf, btag, mask, feats);
    viterbi_kernel<<<1, 704, 0, stream>>>(feats, mask, trans, ptrs, (float*)d_out);
}
